// HybridAtlasBlock_18562848653625
// MI455X (gfx1250) — matrix operand order repaired and checked by reading, not compiled, not run on hardware
//
#include <hip/hip_runtime.h>
#include <math.h>

// ---------------- problem constants (match reference) ----------------
#define D_MODELc 512
#define N_HEADSc 8
#define D_HEADc  64
#define D_PROJc  32
#define SEQ_B    2
#define SEQ_T    2048
#define MROWS    (SEQ_B*SEQ_T)       // 4096

typedef __attribute__((ext_vector_type(16))) _Float16 v16h;
typedef __attribute__((ext_vector_type(8)))  _Float16 v8h;
typedef __attribute__((ext_vector_type(8)))  float    v8f;

// ---------------- small helpers ----------------
__device__ __forceinline__ v8f zero8f(){
  v8f r;
#pragma unroll
  for (int i=0;i<8;++i) r[i]=0.f;
  return r;
}
__device__ __forceinline__ v16h cat16(v8h lo, v8h hi){
  v16h r;
#pragma unroll
  for (int i=0;i<8;++i){ r[i]=lo[i]; r[i+8]=hi[i]; }
  return r;
}
// A fragment (16x32 f16, M x K): row-major LDS tile [rows][lda halfs], lda%8==0
// a[0..7]  = T[row][8*hi .. 8*hi+7] ; a[8..15] = T[row][16+8*hi ..]
__device__ __forceinline__ v16h load_a_frag(const _Float16* base, int row, int lda, int hi){
  const v8h* p0 = (const v8h*)(base + row*lda + 8*hi);
  const v8h* p1 = (const v8h*)(base + row*lda + 16 + 8*hi);
  return cat16(*p0, *p1);
}
// B fragment (32x16 f16, K x N) from K-transposed LDS tile BT[n][lda halfs]:
// b[0..7] = BT[ncol][8*hi .. 8*hi+7] ; b[8..15] = BT[ncol][16+8*hi ..]
__device__ __forceinline__ v16h load_bt_frag(const _Float16* baseT, int ncol, int lda, int hi){
  const v8h* p0 = (const v8h*)(baseT + ncol*lda + 8*hi);
  const v8h* p1 = (const v8h*)(baseT + ncol*lda + 16 + 8*hi);
  return cat16(*p0, *p1);
}
__device__ __forceinline__ v8f wmma16(v16h a, v16h b, v8f c){
  // 8 args: (neg_a, A, neg_b, B, c_mod, C, reuse_a, reuse_b)
  return __builtin_amdgcn_wmma_f32_16x16x32_f16(false, a, false, b, (short)0, c, false, false);
}

// ---- 16-lane butterfly reductions in pure VALU via DPP16 (no LDS round trips) ----
// xor1 = quad_perm[1,0,3,2] (0xB1), xor2 = quad_perm[2,3,0,1] (0x4E),
// xor4 = row_half_mirror (0x141), xor8 = row_mirror (0x140)
template<int CTRL>
__device__ __forceinline__ float dpp_f(float v){
  return __builtin_bit_cast(float,
      __builtin_amdgcn_update_dpp(0, __builtin_bit_cast(int, v), CTRL, 0xf, 0xf, true));
}
__device__ __forceinline__ float grp16_max(float v){
  v = fmaxf(v, dpp_f<0xB1>(v));
  v = fmaxf(v, dpp_f<0x4E>(v));
  v = fmaxf(v, dpp_f<0x141>(v));
  v = fmaxf(v, dpp_f<0x140>(v));
  return v;
}
__device__ __forceinline__ float grp16_sum(float v){
  v += dpp_f<0xB1>(v);
  v += dpp_f<0x4E>(v);
  v += dpp_f<0x141>(v);
  v += dpp_f<0x140>(v);
  return v;
}
__device__ __forceinline__ float wave_sum(float v){
  v = grp16_sum(v);                 // 4 DPP steps within each 16-lane half
  v += __shfl_xor(v, 16, 32);       // single cross-half step
  return v;
}

// ---- CDNA5 async global->LDS copy (ASYNCcnt-tracked), via inline asm ----
// LDS byte address = low 32 bits of the generic pointer.
__device__ __forceinline__ void async_g2l_b128(void* lds_dst, const void* gsrc){
  unsigned ldsb = (unsigned)(unsigned long long)lds_dst;
  asm volatile("global_load_async_to_lds_b128 %0, %1, off"
               :: "v"(ldsb), "v"(gsrc) : "memory");
}
__device__ __forceinline__ void wait_async0(){
  asm volatile("s_wait_asynccnt 0x0" ::: "memory");
}

// ---------------- conversion / weight-prep kernels ----------------
__global__ void k_f32_to_f16(const float* __restrict__ in, _Float16* __restrict__ out, int n){
  int i = blockIdx.x*blockDim.x + threadIdx.x;
  if (i < n) out[i] = (_Float16)in[i];
}

// Fused B matrix for stage-1 GEMM: cols [0,256)=q_projs, [256,512)=k_projs, [512,1024)=w_v^T
__global__ void k_prep_wcat(const float* __restrict__ qp, const float* __restrict__ kp,
                            const float* __restrict__ wv, _Float16* __restrict__ wcat){
  int i = blockIdx.x*blockDim.x + threadIdx.x;     // i = d*1024 + c
  if (i >= D_MODELc*1024) return;
  int d = i >> 10, c = i & 1023;
  float v;
  if (c < 256)      v = qp[(size_t)(c>>5)*(D_MODELc*D_PROJc) + d*D_PROJc + (c&31)];
  else if (c < 512) v = kp[(size_t)((c-256)>>5)*(D_MODELc*D_PROJc) + d*D_PROJc + ((c-256)&31)];
  else              v = wv[(size_t)(c-512)*D_MODELc + d];
  wcat[i] = (_Float16)v;
}

// out[d*N + j] = w[j*512 + d]  (w is (N,512) row-major; out is (512,N) for x @ w.T)
__global__ void k_prep_wT(const float* __restrict__ w, _Float16* __restrict__ out, int N){
  int i = blockIdx.x*blockDim.x + threadIdx.x;
  if (i >= N*D_MODELc) return;
  int d = i / N, j = i % N;
  out[i] = (_Float16)w[(size_t)j*D_MODELc + d];
}

// ---------------- generic WMMA GEMM: C(MxN) = A(MxK,f16) @ B(KxN,f16) + epilogue ----------------
// block = 128 threads = 4 waves (2x2), each wave 32x32 (2x2 WMMA fragments), tile 64x64, K-step 32.
// Double-buffered LDS: tile k+1 staged (A via GLOBAL_LOAD_ASYNC_TO_LDS_B128, B via transpose
// scatter) while tile k is consumed by WMMAs; one s_wait_asynccnt + barrier per K-step.
// EPI: 0 -> Of32 = acc
//      1 -> h = acc + R0 ; Of32 = h ; Of16 = (f16)h            (residual, keep both precisions)
//      2 -> Of32 = R0 + silu(acc)                              (encoder residual branches)
//      3 -> Of32 = tanh(acc / (softplus(R0[col]) + 1e-4))      (final head, N=32)
template<int EPI>
__global__ void __launch_bounds__(128)
k_gemm(const _Float16* __restrict__ A, const _Float16* __restrict__ B,
       int N, int K, const float* __restrict__ R0,
       float* __restrict__ Of32, _Float16* __restrict__ Of16)
{
  __shared__ __align__(16) _Float16 As[2][64*40];   // [row][k], padded rows (80B = 5x16B)
  __shared__ __align__(16) _Float16 Bt[2][64*40];   // [n][k] transposed
  const int tid = threadIdx.x;
  const int wave = tid >> 5, lane = tid & 31, l = lane & 15, hi = lane >> 4;
  const int wm = (wave >> 1) * 32, wn = (wave & 1) * 32;
  const int bm = blockIdx.y * 64,  bn = blockIdx.x * 64;

  auto stage = [&](int k0, int buf){
    // A: 64x32 halfs, async global->LDS 16B copies (2 per thread, fully unrolled)
#pragma unroll
    for (int pass = 0; pass < 2; ++pass){
      int c = tid + pass*128;
      int row = c >> 2, kc = (c & 3) * 8;
      async_g2l_b128(&As[buf][row*40 + kc], A + (size_t)(bm+row)*K + k0 + kc);
    }
    // B transposed: read 8-wide along n (coalesced), scatter into BT[n][k]
#pragma unroll
    for (int pass = 0; pass < 2; ++pass){
      int c = tid + pass*128;
      int kk = c >> 3, nv = (c & 7) * 8;
      v8h t;
      if (bn + nv < N) t = *(const v8h*)(B + (size_t)(k0+kk)*N + bn + nv);
      else {
#pragma unroll
        for (int i=0;i<8;++i) t[i] = (_Float16)0.f;
      }
#pragma unroll
      for (int i=0;i<8;++i) Bt[buf][(nv+i)*40 + kk] = t[i];
    }
  };

  v8f acc[2][2];
#pragma unroll
  for (int i=0;i<2;++i)
#pragma unroll
    for (int j=0;j<2;++j) acc[i][j] = zero8f();

  const int nk = K >> 5;
  stage(0, 0);
  wait_async0();
  __syncthreads();

  for (int kt = 0; kt < nk; ++kt){
    const int buf = kt & 1;
    if (kt + 1 < nk) stage((kt+1)*32, buf ^ 1);   // copy next tile under this tile's WMMAs
    v16h af[2], bf[2];
#pragma unroll
    for (int i=0;i<2;++i) af[i] = load_a_frag(&As[buf][0], wm + 16*i + l, 40, hi);
#pragma unroll
    for (int j=0;j<2;++j) bf[j] = load_bt_frag(&Bt[buf][0], wn + 16*j + l, 40, hi);
#pragma unroll
    for (int i=0;i<2;++i)
#pragma unroll
      for (int j=0;j<2;++j)
        acc[i][j] = wmma16(af[i], bf[j], acc[i][j]);
    wait_async0();          // my async writes for tile kt+1 have landed
    __syncthreads();        // publish buf^1; everyone done reading buf
  }

  // epilogue + store (C layout: VGPR r -> row r+8*hi, lane -> col)
#pragma unroll
  for (int i=0;i<2;++i)
#pragma unroll
    for (int j=0;j<2;++j){
      int col = bn + wn + 16*j + l;
      if (col >= N) continue;
#pragma unroll
      for (int r=0;r<8;++r){
        int row = bm + wm + 16*i + r + 8*hi;
        size_t idx = (size_t)row * N + col;
        float v = acc[i][j][r];
        if (EPI == 0){
          Of32[idx] = v;
        } else if (EPI == 1){
          float h = v + R0[idx];
          Of32[idx] = h;
          Of16[idx] = (_Float16)h;
        } else if (EPI == 2){
          float s = v / (1.f + __expf(-v));
          Of32[idx] = R0[idx] + s;
        } else {
          float sp = __logf(1.f + __expf(R0[col])) + 1e-4f;
          Of32[idx] = tanhf(v / sp);
        }
      }
    }
}

// ---------------- RoPE + gain + score-scale pack: qkv f32 -> qf/kf/vf (f16) + v f32 ----------------
__global__ void k_rope_pack(const float* __restrict__ qkv, const float* __restrict__ qgain,
                            _Float16* __restrict__ qf, _Float16* __restrict__ kf,
                            _Float16* __restrict__ vf, float* __restrict__ v32)
{
  int i = blockIdx.x*blockDim.x + threadIdx.x;   // one (b,h,t) per thread
  if (i >= SEQ_B*N_HEADSc*SEQ_T) return;
  int t  = i % SEQ_T;
  int bh = i / SEQ_T;
  int h  = bh % N_HEADSc;
  int b  = bh / N_HEADSc;
  const float* row = qkv + (size_t)(b*SEQ_T + t) * 1024;
  const float gscale = qgain[h] * 0.17677669529663687f;   // q_gain * 1/sqrt(32) folded into Q
  const float lnf    = -0.5756462732485114f;              // -ln(10000)/16
  _Float16* qo = qf + (size_t)(bh*SEQ_T + t)*D_PROJc;
  _Float16* ko = kf + (size_t)(bh*SEQ_T + t)*D_PROJc;
#pragma unroll
  for (int ii=0; ii<16; ++ii){
    float ang = __expf(lnf * (float)ii) * (float)t;
    float s, c;
    __sincosf(ang, &s, &c);
    float q1 = row[h*32 + ii],       q2 = row[h*32 + 16 + ii];
    float k1 = row[256 + h*32 + ii], k2 = row[256 + h*32 + 16 + ii];
    qo[ii]    = (_Float16)((q1*c - q2*s) * gscale);
    qo[16+ii] = (_Float16)((q2*c + q1*s) * gscale);
    ko[ii]    = (_Float16)(k1*c - k2*s);
    ko[16+ii] = (_Float16)(k2*c + k1*s);
  }
  _Float16* vo = vf  + (size_t)(bh*SEQ_T + t)*D_HEADc;
  float*    v3 = v32 + (size_t)(bh*SEQ_T + t)*D_HEADc;
#pragma unroll
  for (int d=0; d<64; ++d){
    float v = row[512 + h*64 + d];
    vo[d] = (_Float16)v;
    v3[d] = v;
  }
}

// ---------------- flash attention (causal) + self-align epilogue ----------------
// grid = (T/64, B*H), block = 128 (4 waves). Each wave owns a 16-query tile; the block shares
// double-buffered K/V staging for 32-key chunks (K tile via async global->LDS), staged one
// tile ahead of the WMMAs. Online softmax fully in VALU (DPP16), mask-free fast path for
// interior tiles. The per-wave Plds relayout needs no barrier (same-wave LDS is in-order).
__global__ void __launch_bounds__(128)
k_flash(const _Float16* __restrict__ qf, const _Float16* __restrict__ kf,
        const _Float16* __restrict__ vf, const float* __restrict__ v32,
        _Float16* __restrict__ yf)
{
  __shared__ __align__(16) _Float16 Klds[2][32*40];   // [key][d]
  __shared__ __align__(16) _Float16 Vt[2][64*40];     // [dh][key] (transposed)
  __shared__ __align__(16) _Float16 Plds[4][16*40];   // per-wave P tile for A-fragment relayout
  const int tid = threadIdx.x, wave = tid>>5, lane = tid&31, l = lane&15, hi = lane>>4;
  const int bh = blockIdx.y;
  const int b  = bh >> 3, h = bh & 7;
  const int qt = blockIdx.x*4 + wave;                 // 16-row query tile index
  const size_t rowbase = (size_t)bh * SEQ_T;

  auto stage = [&](int kt, int buf){
    const int s0 = kt * 32;
    {   // K tile: 32 keys x 32 d, one async 16B copy per thread (128 chunks)
      int key = tid >> 2, dv = (tid & 3) * 8;
      async_g2l_b128(&Klds[buf][key*40 + dv], kf + (rowbase + s0 + key)*D_PROJc + dv);
    }
#pragma unroll
    for (int pass = 0; pass < 2; ++pass){   // V transposed: Vt[dh][key]
      int c = tid + pass*128;
      int key = c >> 3, dv = (c & 7) * 8;
      v8h t = *(const v8h*)(vf + (rowbase + s0 + key)*D_HEADc + dv);
#pragma unroll
      for (int i=0;i<8;++i) Vt[buf][(dv+i)*40 + key] = t[i];
    }
  };

  // persistent Q A-fragment (scale+gain already folded in)
  v16h qa;
  {
    const _Float16* qrow = qf + (rowbase + qt*16 + l)*D_PROJc;
    qa = cat16(*(const v8h*)(qrow + 8*hi), *(const v8h*)(qrow + 16 + 8*hi));
  }
  v8f acc[4];
#pragma unroll
  for (int nt=0; nt<4; ++nt) acc[nt] = zero8f();
  float mrow[8], lrow[8];
#pragma unroll
  for (int r=0;r<8;++r){ mrow[r] = -1e30f; lrow[r] = 0.f; }

  const int my_ktmax  = (qt*16 + 15) >> 5;
  const int blk_ktmax = ((blockIdx.x*4 + 3)*16 + 15) >> 5;

  stage(0, 0);
  wait_async0();
  __syncthreads();

  for (int kt = 0; kt <= blk_ktmax; ++kt){
    const int s0 = kt * 32;
    const int buf = kt & 1;
    if (kt < blk_ktmax) stage(kt+1, buf ^ 1);   // copy next tile under this tile's compute

    if (kt <= my_ktmax){
      v16h b0 = load_bt_frag(&Klds[buf][0], l,      40, hi);   // keys s0..s0+15
      v16h b1 = load_bt_frag(&Klds[buf][0], 16 + l, 40, hi);   // keys s0+16..s0+31
      v8f c0 = zero8f(), c1 = zero8f();
      c0 = wmma16(qa, b0, c0);
      c1 = wmma16(qa, b1, c1);
      if (s0 + 31 <= qt*16){
        // interior tile: no causal masking needed (all keys <= all query rows)
#pragma unroll
        for (int r=0;r<8;++r){
          float rm = grp16_max(fmaxf(c0[r], c1[r]));
          float mn = fmaxf(mrow[r], rm);
          float sc = __expf(mrow[r] - mn);
          mrow[r] = mn;
          float p0 = __expf(c0[r] - mn);
          float p1 = __expf(c1[r] - mn);
          lrow[r] = lrow[r]*sc + grp16_sum(p0 + p1);
#pragma unroll
          for (int nt=0; nt<4; ++nt) acc[nt][r] *= sc;
          Plds[wave][(r + 8*hi)*40 + l]      = (_Float16)p0;
          Plds[wave][(r + 8*hi)*40 + 16 + l] = (_Float16)p1;
        }
      } else {
        // diagonal tile: apply causal mask
#pragma unroll
        for (int r=0;r<8;++r){
          const int qrow = qt*16 + r + 8*hi;
          float s0v = (s0 + l      <= qrow) ? c0[r] : -1e30f;
          float s1v = (s0 + 16 + l <= qrow) ? c1[r] : -1e30f;
          float rm = grp16_max(fmaxf(s0v, s1v));
          float mn = fmaxf(mrow[r], rm);
          float sc = __expf(mrow[r] - mn);
          mrow[r] = mn;
          float p0 = __expf(s0v - mn);
          float p1 = __expf(s1v - mn);
          lrow[r] = lrow[r]*sc + grp16_sum(p0 + p1);
#pragma unroll
          for (int nt=0; nt<4; ++nt) acc[nt][r] *= sc;
          Plds[wave][(r + 8*hi)*40 + l]      = (_Float16)p0;
          Plds[wave][(r + 8*hi)*40 + 16 + l] = (_Float16)p1;
        }
      }
      // same-wave LDS RAW: DS ops are in-order within a wave; no barrier needed.
      // Preload P and all 4 V fragments, then issue the 4 WMMAs back-to-back so the
      // matrix pipe is not stalled on per-op dscnt waits.
      v16h pa = load_a_frag(&Plds[wave][0], l, 40, hi);
      v16h vb[4];
#pragma unroll
      for (int nt=0; nt<4; ++nt) vb[nt] = load_bt_frag(&Vt[buf][0], nt*16 + l, 40, hi);
#pragma unroll
      for (int nt=0; nt<4; ++nt) acc[nt] = wmma16(pa, vb[nt], acc[nt]);
    }
    wait_async0();          // my async writes for tile kt+1 have landed
    __syncthreads();        // publish buf^1; everyone done reading buf
  }

  // epilogue: 1/lsum, then y -= (y . v_hat) v_hat with v_hat = v/max(||v||,1e-12)
#pragma unroll
  for (int r=0;r<8;++r){
    const int tg = qt*16 + r + 8*hi;
    const float inv = 1.f / lrow[r];
    const float* vr = v32 + (rowbase + tg)*D_HEADc;
    float yv[4], vv[4], s1 = 0.f, s2 = 0.f;
#pragma unroll
    for (int nt=0; nt<4; ++nt){
      yv[nt] = acc[nt][r] * inv;
      vv[nt] = vr[nt*16 + l];
      s1 += yv[nt]*vv[nt];
      s2 += vv[nt]*vv[nt];
    }
    s1 = grp16_sum(s1);
    s2 = grp16_sum(s2);
    float invn = 1.f / fmaxf(sqrtf(s2), 1e-12f);
    float coef = s1 * invn * invn;
    _Float16* orow = yf + (size_t)(b*SEQ_T + tg)*D_MODELc + h*D_HEADc;
#pragma unroll
    for (int nt=0; nt<4; ++nt)
      orow[nt*16 + l] = (_Float16)(yv[nt] - coef*vv[nt]);
  }
}

// ---------------- LayerNorm / RMSNorm rows of 512 (one wave per row, 8 rows per block) ----------------
__global__ void __launch_bounds__(256)
k_layernorm(const float* __restrict__ in, const float* __restrict__ g,
            const float* __restrict__ bb, _Float16* __restrict__ out)
{
  int wave = threadIdx.x >> 5, lane = threadIdx.x & 31;
  int row = blockIdx.x*8 + wave;
  const float* r = in + (size_t)row*D_MODELc;
  float vals[16], s = 0.f, sq = 0.f;
#pragma unroll
  for (int i=0;i<16;++i){
    float v = r[lane + i*32];
    vals[i] = v; s += v; sq += v*v;
  }
  s = wave_sum(s); sq = wave_sum(sq);
  float mean = s * (1.f/512.f);
  float var  = sq * (1.f/512.f) - mean*mean;
  float rstd = rsqrtf(var + 1e-5f);
  _Float16* o = out + (size_t)row*D_MODELc;
#pragma unroll
  for (int i=0;i<16;++i){
    int col = lane + i*32;
    o[col] = (_Float16)((vals[i]-mean)*rstd*g[col] + bb[col]);
  }
}

__global__ void __launch_bounds__(256)
k_rmsnorm(const float* __restrict__ in, _Float16* __restrict__ out)
{
  int wave = threadIdx.x >> 5, lane = threadIdx.x & 31;
  int row = blockIdx.x*8 + wave;
  const float* r = in + (size_t)row*D_MODELc;
  float vals[16], sq = 0.f;
#pragma unroll
  for (int i=0;i<16;++i){
    float v = r[lane + i*32];
    vals[i] = v; sq += v*v;
  }
  sq = wave_sum(sq);
  float rms = rsqrtf(sq * (1.f/512.f) + 1e-6f);
  _Float16* o = out + (size_t)row*D_MODELc;
#pragma unroll
  for (int i=0;i<16;++i)
    o[lane + i*32] = (_Float16)(vals[i]*rms);
}

// ---------------- host-side orchestration ----------------
extern "C" void kernel_launch(void* const* d_in, const int* in_sizes, int n_in,
                              void* d_out, int out_size, void* d_ws, size_t ws_size,
                              hipStream_t stream) {
  (void)in_sizes; (void)n_in; (void)out_size; (void)ws_size;
  const float* x   = (const float*)d_in[0];
  const float* qp  = (const float*)d_in[1];
  const float* kp  = (const float*)d_in[2];
  const float* wv  = (const float*)d_in[3];
  const float* wpr = (const float*)d_in[4];
  const float* qg  = (const float*)d_in[5];
  const float* w0  = (const float*)d_in[6];
  const float* g1  = (const float*)d_in[7];
  const float* b1  = (const float*)d_in[8];
  const float* w1  = (const float*)d_in[9];
  const float* g2  = (const float*)d_in[10];
  const float* b2  = (const float*)d_in[11];
  const float* w2  = (const float*)d_in[12];
  const float* wo  = (const float*)d_in[13];
  const float* ct  = (const float*)d_in[14];
  float* zout = (float*)d_out;

  char* ws = (char*)d_ws;
  size_t off = 0;
  auto alloc = [&](size_t bytes)->void*{
    void* p = ws + off;
    off = (off + bytes + 255) & ~(size_t)255;
    return p;
  };
  _Float16* xf16 = (_Float16*)alloc((size_t)MROWS*512*2);
  _Float16* wcat = (_Float16*)alloc((size_t)512*1024*2);
  _Float16* wprT = (_Float16*)alloc((size_t)512*512*2);
  _Float16* w0T  = (_Float16*)alloc((size_t)512*512*2);
  _Float16* w1T  = (_Float16*)alloc((size_t)512*512*2);
  _Float16* w2T  = (_Float16*)alloc((size_t)512*512*2);
  _Float16* woT  = (_Float16*)alloc((size_t)512*32*2);
  float*    qkv  = (float*)  alloc((size_t)MROWS*1024*4);
  _Float16* qf   = (_Float16*)alloc((size_t)16*SEQ_T*32*2);
  _Float16* kff  = (_Float16*)alloc((size_t)16*SEQ_T*32*2);
  _Float16* vf   = (_Float16*)alloc((size_t)16*SEQ_T*64*2);
  float*    v32  = (float*)  alloc((size_t)16*SEQ_T*64*4);
  _Float16* yf   = (_Float16*)alloc((size_t)MROWS*512*2);
  float*    hf   = (float*)  alloc((size_t)MROWS*512*4);
  _Float16* hf16 = (_Float16*)alloc((size_t)MROWS*512*2);
  float*    he0  = (float*)  alloc((size_t)MROWS*512*4);
  _Float16* ln0  = (_Float16*)alloc((size_t)MROWS*512*2);
  float*    he1  = (float*)  alloc((size_t)MROWS*512*4);
  _Float16* ln1  = (_Float16*)alloc((size_t)MROWS*512*2);
  float*    he2  = (float*)  alloc((size_t)MROWS*512*4);
  _Float16* hr   = (_Float16*)alloc((size_t)MROWS*512*2);

  // weight / activation prep
  k_f32_to_f16<<<(MROWS*512+255)/256, 256, 0, stream>>>(x, xf16, MROWS*512);
  k_prep_wcat<<<(512*1024+255)/256, 256, 0, stream>>>(qp, kp, wv, wcat);
  k_prep_wT<<<(512*512+255)/256, 256, 0, stream>>>(wpr, wprT, 512);
  k_prep_wT<<<(512*512+255)/256, 256, 0, stream>>>(w0,  w0T,  512);
  k_prep_wT<<<(512*512+255)/256, 256, 0, stream>>>(w1,  w1T,  512);
  k_prep_wT<<<(512*512+255)/256, 256, 0, stream>>>(w2,  w2T,  512);
  k_prep_wT<<<(32*512+255)/256, 256, 0, stream>>>(wo, woT, 32);

  // 1) fused q/k/v projection GEMM: (4096x512) @ (512x1024)
  k_gemm<0><<<dim3(1024/64, MROWS/64), 128, 0, stream>>>(xf16, wcat, 1024, 512, nullptr, qkv, nullptr);
  // 2) RoPE + gain + 1/sqrt(d) fold; pack f16 q/k/v (+ f32 v for self-align)
  k_rope_pack<<<(SEQ_B*N_HEADSc*SEQ_T+255)/256, 256, 0, stream>>>(qkv, qg, qf, kff, vf, v32);
  // 3) causal flash attention + self-align removal -> y (f16, (B*T)x512)
  k_flash<<<dim3(SEQ_T/64, SEQ_B*N_HEADSc), 128, 0, stream>>>(qf, kff, vf, v32, yf);
  // 4) h = x + y @ w_proj.T
  k_gemm<1><<<dim3(512/64, MROWS/64), 128, 0, stream>>>(yf, wprT, 512, 512, x, hf, hf16);
  // 5) encoder: he0 = h @ enc_w0.T ; he1 = h + silu(LN(he0) @ enc_w1.T) ; he2 = he1 + silu(LN(he1) @ enc_w2.T)
  k_gemm<0><<<dim3(512/64, MROWS/64), 128, 0, stream>>>(hf16, w0T, 512, 512, nullptr, he0, nullptr);
  k_layernorm<<<MROWS/8, 256, 0, stream>>>(he0, g1, b1, ln0);
  k_gemm<2><<<dim3(512/64, MROWS/64), 128, 0, stream>>>(ln0, w1T, 512, 512, hf, he1, nullptr);
  k_layernorm<<<MROWS/8, 256, 0, stream>>>(he1, g2, b2, ln1);
  k_gemm<2><<<dim3(512/64, MROWS/64), 128, 0, stream>>>(ln1, w2T, 512, 512, he1, he2, nullptr);
  // 6) RMS norm + head: z = tanh((h_rms @ enc_w_out.T) / (softplus(coord_temp)+1e-4))
  k_rmsnorm<<<MROWS/8, 256, 0, stream>>>(he2, hr);
  k_gemm<3><<<dim3(1, MROWS/64), 128, 0, stream>>>(hr, woT, 32, 512, ct, zout, nullptr);
}
